// GraphAggregator_54288386621833
// MI455X (gfx1250) — matrix operand order repaired and checked by reading, not compiled, not run on hardware
//
#include <hip/hip_runtime.h>
#include <hip/hip_bf16.h>

// GraphRec attention aggregation for MI455X (gfx1250, wave32).
// One wave per node. MLP via V_WMMA_F32_16X16X32_BF16 (f32 accumulate).
// Embedding table (128MB) is L2-resident (192MB L2) -> gathers are cheap;
// kernel is compute-bound on the per-pair MLP, which we map to WMMA tiles.

typedef __attribute__((ext_vector_type(16))) __bf16 v16bf;
typedef __attribute__((ext_vector_type(8)))  __bf16 v8bf;
typedef __attribute__((ext_vector_type(8)))  float  v8f;

#define NW   2      // waves (nodes) per block -> 64 threads, ~58KB static LDS
#define KNEI 32     // neighbors per node (compile-time: WMMA tiling depends on it)
#define DIM  64     // embedding dim
#define H2S  68     // padded H2 row stride (floats) to dodge LDS bank conflicts

static __device__ inline v8bf ld8(const __bf16* p) { return *(const v8bf*)p; }

static __device__ inline v16bf mk16(v8bf lo, v8bf hi) {
  return __builtin_shufflevector(lo, hi, 0,1,2,3,4,5,6,7,8,9,10,11,12,13,14,15);
}

// A-fragment (16x32 bf16): lane<16 holds K[0..8)+K[16..24), lane>=16 K[8..16)+K[24..32)
// rowbase must already include the +8*half element offset.
static __device__ inline v16bf loadA(const __bf16* rowbase) {
  return mk16(ld8(rowbase), ld8(rowbase + 16));
}

static __device__ inline v8f wmma_bf16(v16bf a, v16bf b, v8f c) {
  return __builtin_amdgcn_wmma_f32_16x16x32_bf16(false, a, false, b, (short)0, c,
                                                 false, false);
}

__global__ __launch_bounds__(NW * 32, 1)
void graphrec_attn_kernel(const float* __restrict__ emb,
                          const float* __restrict__ W1,
                          const float* __restrict__ b1,
                          const float* __restrict__ W2,
                          const float* __restrict__ b2,
                          const float* __restrict__ w3,
                          const int*   __restrict__ video_nodes,
                          const int*   __restrict__ neigh_idx,
                          const int*   __restrict__ neigh_deg,
                          float*       __restrict__ out,
                          int N) {
  // Block-shared (staged once): transposed bf16 weights so a B-fragment
  // (two runs of 8 K values for one output column) is two 16B ds loads.
  __shared__ __align__(32) __bf16 sW1T[64 * 128];   // [n][k] 16KB
  __shared__ __align__(32) __bf16 sW2T[64 * 64];    // [n][k]  8KB
  __shared__ float sw3[64], sb1[64], sb2[64];
  // Per-wave scratch:
  __shared__ __align__(32) __bf16 sXe[NW][KNEI * DIM];  // neighbor embeds (bf16)
  __shared__ __align__(32) __bf16 sU [NW][DIM];         // center embed (bf16)
  __shared__ __align__(32) __bf16 sH1[NW][KNEI * DIM];  // relu(X@W1+b1) bf16
  __shared__ __align__(16) float  sH2[NW][KNEI * H2S];  // relu(H1@W2+b2) f32

  const int tid = threadIdx.x;

  // ---- Stage weights (transpose + convert to bf16), once per block ----
  #pragma unroll 4
  for (int i = tid; i < 128 * 64; i += NW * 32)
    sW1T[(i & 63) * 128 + (i >> 6)] = (__bf16)W1[i];
  #pragma unroll 2
  for (int i = tid; i < 64 * 64; i += NW * 32)
    sW2T[(i & 63) * 64 + (i >> 6)] = (__bf16)W2[i];
  for (int i = tid; i < 64; i += NW * 32) {
    sw3[i] = w3[i]; sb1[i] = b1[i]; sb2[i] = b2[i];
  }
  __syncthreads();

  const int wv   = tid >> 5;
  const int l    = tid & 31;   // lane
  const int lm   = l & 15;
  const int half = l >> 4;     // 0: lanes 0-15, 1: lanes 16-31
  const int node = blockIdx.x * NW + wv;
  if (node >= N) return;       // wave-uniform; no barriers past this point

  const int deg    = neigh_deg[node];
  const int center = video_nodes[node];
  const int nb     = neigh_idx[node * KNEI + l];   // lane l owns neighbor l

  __bf16* Xe = sXe[wv];
  __bf16* U  = sU[wv];
  __bf16* H1 = sH1[wv];
  float*  H2 = sH2[wv];

  // ---- Gather center + neighbor rows -> bf16 LDS (coalesced, L2-resident) ----
  {
    float2 uv = *(const float2*)(emb + (long)center * DIM + 2 * l);
    U[2 * l] = (__bf16)uv.x; U[2 * l + 1] = (__bf16)uv.y;
  }
  for (int k = 0; k < KNEI; ++k) {
    int id = __shfl(nb, k, 32);
    float2 ev = *(const float2*)(emb + (long)id * DIM + 2 * l);
    Xe[k * DIM + 2 * l]     = (__bf16)ev.x;
    Xe[k * DIM + 2 * l + 1] = (__bf16)ev.y;
  }
  __builtin_amdgcn_wave_barrier();   // LDS is in-order within a wave

  // ---- GEMM1: X[32,128] @ W1[128,64]  (X = [e | u]) ----
  v8f C1[2][4];
  {
    v8f z = {0.f,0.f,0.f,0.f,0.f,0.f,0.f,0.f};
    #pragma unroll
    for (int m = 0; m < 2; ++m)
      #pragma unroll
      for (int n = 0; n < 4; ++n) C1[m][n] = z;
  }
  #pragma unroll
  for (int kc = 0; kc < 4; ++kc) {       // 4 K-chunks of 32
    v16bf a0, a1;
    if (kc < 2) {                        // columns 0..63: neighbor part
      const __bf16* p0 = Xe + lm * DIM + kc * 32 + half * 8;
      const __bf16* p1 = Xe + (lm + 16) * DIM + kc * 32 + half * 8;
      a0 = loadA(p0);
      a1 = loadA(p1);
    } else {                             // columns 64..127: center u, same for all rows
      const __bf16* pu = U + (kc - 2) * 32 + half * 8;
      a0 = loadA(pu);
      a1 = a0;                           // identical for both M-tiles
    }
    #pragma unroll
    for (int nt = 0; nt < 4; ++nt) {
      const __bf16* pb = sW1T + (nt * 16 + lm) * 128 + kc * 32 + half * 8;
      v16bf b = mk16(ld8(pb), ld8(pb + 16));
      C1[0][nt] = wmma_bf16(a0, b, C1[0][nt]);
      C1[1][nt] = wmma_bf16(a1, b, C1[1][nt]);
    }
  }

  // ---- bias + relu -> H1 (bf16, row-major for next A fragments) ----
  #pragma unroll
  for (int nt = 0; nt < 4; ++nt) {
    float bv = sb1[nt * 16 + lm];
    #pragma unroll
    for (int m = 0; m < 2; ++m)
      #pragma unroll
      for (int r = 0; r < 8; ++r) {
        float h = fmaxf(C1[m][nt][r] + bv, 0.0f);
        H1[(m * 16 + half * 8 + r) * DIM + nt * 16 + lm] = (__bf16)h;
      }
  }
  __builtin_amdgcn_wave_barrier();

  // ---- GEMM2: H1[32,64] @ W2[64,64] ----
  v8f C2[2][4];
  {
    v8f z = {0.f,0.f,0.f,0.f,0.f,0.f,0.f,0.f};
    #pragma unroll
    for (int m = 0; m < 2; ++m)
      #pragma unroll
      for (int n = 0; n < 4; ++n) C2[m][n] = z;
  }
  #pragma unroll
  for (int kc = 0; kc < 2; ++kc) {
    const __bf16* p0 = H1 + lm * DIM + kc * 32 + half * 8;
    const __bf16* p1 = H1 + (lm + 16) * DIM + kc * 32 + half * 8;
    v16bf a0 = loadA(p0);
    v16bf a1 = loadA(p1);
    #pragma unroll
    for (int nt = 0; nt < 4; ++nt) {
      const __bf16* pb = sW2T + (nt * 16 + lm) * 64 + kc * 32 + half * 8;
      v16bf b = mk16(ld8(pb), ld8(pb + 16));
      C2[0][nt] = wmma_bf16(a0, b, C2[0][nt]);
      C2[1][nt] = wmma_bf16(a1, b, C2[1][nt]);
    }
  }

  // ---- bias + relu -> H2 (f32, padded rows) ----
  #pragma unroll
  for (int nt = 0; nt < 4; ++nt) {
    float bv = sb2[nt * 16 + lm];
    #pragma unroll
    for (int m = 0; m < 2; ++m)
      #pragma unroll
      for (int r = 0; r < 8; ++r) {
        float h = fmaxf(C2[m][nt][r] + bv, 0.0f);
        H2[(m * 16 + half * 8 + r) * H2S + nt * 16 + lm] = h;
      }
  }
  __builtin_amdgcn_wave_barrier();

  // ---- scores: lane k owns neighbor k; s[k] = H2[k,:] . w3  ----
  // (b3 is a uniform shift -> softmax-invariant, safely omitted)
  float s = 0.0f;
  const float* hrow = H2 + l * H2S;
  #pragma unroll
  for (int i = 0; i < 16; ++i) {
    float4 v = *(const float4*)(hrow + i * 4);
    s += v.x * sw3[4 * i] + v.y * sw3[4 * i + 1] +
         v.z * sw3[4 * i + 2] + v.w * sw3[4 * i + 3];
  }
  const bool valid = (l < deg);
  s = valid ? s : -1e30f;

  // ---- wave-wide masked softmax (wave32 shfl reductions) ----
  float mx = s;
  #pragma unroll
  for (int off = 16; off >= 1; off >>= 1) mx = fmaxf(mx, __shfl_xor(mx, off, 32));
  float ex = valid ? __expf(s - mx) : 0.0f;
  float sum = ex;
  #pragma unroll
  for (int off = 16; off >= 1; off >>= 1) sum += __shfl_xor(sum, off, 32);

  // ---- attention-weighted sum of fp32 neighbor embeddings ----
  float2 res;
  if (deg > 0) {                       // wave-uniform branch
    float att = ex / sum;
    float2 acc = {0.0f, 0.0f};
    for (int k = 0; k < KNEI; ++k) {
      float a  = __shfl(att, k, 32);
      int   id = __shfl(nb,  k, 32);
      float2 ev = *(const float2*)(emb + (long)id * DIM + 2 * l);  // L2 hit
      acc.x += a * ev.x; acc.y += a * ev.y;
    }
    res = acc;
  } else {                             // no neighbors -> node's own embedding
    res = *(const float2*)(emb + (long)center * DIM + 2 * l);
  }
  *(float2*)(out + (long)node * DIM + 2 * l) = res;
}

extern "C" void kernel_launch(void* const* d_in, const int* in_sizes, int n_in,
                              void* d_out, int out_size, void* d_ws, size_t ws_size,
                              hipStream_t stream) {
  const float* emb = (const float*)d_in[0];
  const float* W1  = (const float*)d_in[1];
  const float* b1  = (const float*)d_in[2];
  const float* W2  = (const float*)d_in[3];
  const float* b2  = (const float*)d_in[4];
  const float* w3  = (const float*)d_in[5];
  // d_in[6] = b3: uniform additive shift before softmax -> no effect, unused.
  const int* vn   = (const int*)d_in[7];
  const int* nidx = (const int*)d_in[8];
  const int* ndeg = (const int*)d_in[9];
  float* out = (float*)d_out;

  const int N = in_sizes[7];                 // video_nodes count
  const int blocks = (N + NW - 1) / NW;
  graphrec_attn_kernel<<<blocks, NW * 32, 0, stream>>>(
      emb, W1, b1, W2, b2, w3, vn, nidx, ndeg, out, N);
}
